// RecurrentGCN_7756710936770
// MI455X (gfx1250) — compile-verified, matrix operand order checked by reading, not run on hardware
//
#include <hip/hip_runtime.h>
#include <hip/hip_bf16.h>
#include <cstddef>

// ---------------------------------------------------------------------------
// RecurrentGCN (5 stacked GConvGRU, K=1 => per-node linears) on gfx1250.
// Per-node recurrence is independent: each wave owns 16 nodes, keeps all 5
// hidden states in LDS across T=8 steps, and runs every GEMM through
// v_wmma_f32_16x16x32_f16 (f16 A/B, f32 accumulate).
// ---------------------------------------------------------------------------

typedef _Float16 h16 __attribute__((ext_vector_type(16)));
typedef _Float16 h8v __attribute__((ext_vector_type(8)));
typedef float    f8  __attribute__((ext_vector_type(8)));

static constexpr int DIN_[5]  = {64, 256, 128, 64, 32};
static constexpr int DOUT_[5] = {256, 128, 64, 32, 2};
static constexpr int NPAD_[5] = {256, 128, 64, 32, 16};   // dout padded to 16
static constexpr int KHP_[5]  = {256, 128, 64, 32, 32};   // dout padded to 32 (Wh K-dim)
static constexpr int HOFF_[5] = {0, 256, 384, 448, 480};  // feature offset of each layer's H
static constexpr int HSTRIDE  = 496;                      // f16 per node (16B aligned rows)
static constexpr int LBASE_[5]= {0, 245760, 393216, 430080, 439296};
static constexpr int WTOTAL   = 442368;                   // f16 elements of packed weights
static constexpr int T_STEPS  = 8;
static constexpr int NWAVES   = 2;                        // waves per block (independent)

// packed-weight offset; matrix order per layer: [Wxz, Whz, Wxr, Whr, Wxh, Whh]
__host__ __device__ constexpr int woff(int L, int mi) {
    int wx = DIN_[L] * NPAD_[L];
    int wh = KHP_[L] * NPAD_[L];
    int o  = (mi == 0) ? 0
           : (mi == 1) ? wx
           : (mi == 2) ? wx + wh
           : (mi == 3) ? 2 * wx + wh
           : (mi == 4) ? 2 * wx + 2 * wh
                       : 3 * wx + 2 * wh;
    return LBASE_[L] + o;
}

struct Params {
    const float* graphs;        // [T, N, 64] fp32
    float*       out;           // [T, N, 2]  fp32
    const _Float16* wts;        // packed f16 WMMA B-tiles in workspace
    const float* bias[30];      // [layer*6 + {xz,hz,xr,hr,xh,hh}]
    int N;
};

// --- fp32 (K,N) row-major -> f16 WMMA B-tile layout, zero-padded -----------
// Tile (kt,nt) of 32x16 stored as 512 f16 at tile index nt*KT+kt; within a
// tile, element (kk,nn): lane = nn + 16*((kk>>3)&1), slot = (kk&7) + 8*(kk>>4).
__global__ void repack_w(const float* __restrict__ src, _Float16* __restrict__ dst,
                         int K, int Nact, int Kpad, int Np) {
    int gid = blockIdx.x * blockDim.x + threadIdx.x;
    int total = Kpad * Np;
    if (gid >= total) return;
    int k = gid / Np;
    int n = gid - k * Np;
    float v = (k < K && n < Nact) ? src[k * Nact + n] : 0.0f;
    int kt = k >> 5, nt = n >> 4, kk = k & 31, nn = n & 15;
    int lane = nn + 16 * ((kk >> 3) & 1);
    int slot = (kk & 7) + 8 * (kk >> 4);
    int ktiles = Kpad >> 5;
    dst[(size_t)(nt * ktiles + kt) * 512 + lane * 16 + slot] = (_Float16)v;
}

// --- fragment helpers -------------------------------------------------------
static __device__ inline f8 wmma_f16(h16 a, h16 b, f8 c) {
    return __builtin_amdgcn_wmma_f32_16x16x32_f16(false, a, false, b, (short)0, c,
                                                  false, false);
}

static __device__ inline h16 load_b(const _Float16* __restrict__ w,
                                    int nt, int kt, int KT, int lane) {
    return *(const h16*)(w + (size_t)(nt * KT + kt) * 512 + lane * 16);
}

// 16-bit A layout: lane<16 -> row m=lane, K {0..7,16..23}; lane>=16 -> K {8..15,24..31}
static __device__ inline h16 load_a_lds(const _Float16* __restrict__ buf, int stride,
                                        int foff, int kt, int limit, int lane) {
    int m = lane & 15, hf = lane >> 4;
    int f0 = kt * 32 + hf * 8;
    int f1 = f0 + 16;
    const _Float16* base = buf + m * stride + foff;
    h8v z = {};
    h8v c0 = (f0 < limit) ? *(const h8v*)(base + f0) : z;
    h8v c1 = (f1 < limit) ? *(const h8v*)(base + f1) : z;
    return __builtin_shufflevector(c0, c1, 0, 1, 2, 3, 4, 5, 6, 7,
                                   8, 9, 10, 11, 12, 13, 14, 15);
}

template <int L>
static __device__ inline h16 load_x_frag(const float* __restrict__ xg,
                                         const _Float16* __restrict__ Hw,
                                         int nodeBase, int Ntot, int lane, int kt) {
    if constexpr (L == 0) {
        int m = lane & 15, hf = lane >> 4;
        int node = nodeBase + m;
        if (node >= Ntot) node = Ntot - 1;             // clamp: dup work, no EXEC change
        const float* p = xg + (size_t)node * 64 + kt * 32 + hf * 8;
        h16 r;
#pragma unroll
        for (int i = 0; i < 8; ++i) r[i] = (_Float16)p[i];
#pragma unroll
        for (int i = 0; i < 8; ++i) r[8 + i] = (_Float16)p[16 + i];
        return r;
    } else {
        constexpr int xoff = HOFF_[L - 1];
        constexpr int din  = DIN_[L];
        return load_a_lds(Hw, HSTRIDE, xoff, kt, din, lane);
    }
}

static __device__ inline float sigmoidf_(float x) {
    return 1.0f / (1.0f + __expf(-x));
}

// --- one GRU layer for one timestep (one wave = 16 nodes) -------------------
template <int L>
static __device__ void layer_step(const Params& p, const float* __restrict__ xg,
                                  _Float16* __restrict__ Hw,
                                  _Float16* __restrict__ RHw,
                                  _Float16* __restrict__ HNw,
                                  int nodeBase, int lane) {
    constexpr int din  = DIN_[L];
    constexpr int dout = DOUT_[L];
    constexpr int NP   = NPAD_[L];
    constexpr int KHp  = KHP_[L];
    constexpr int NT   = NP / 16;
    constexpr int KXT  = din / 32;
    constexpr int KHT  = KHp / 32;
    constexpr int hoff = HOFF_[L];

    constexpr int oxz = woff(L, 0), ohz = woff(L, 1), oxr = woff(L, 2);
    constexpr int ohr = woff(L, 3), oxh = woff(L, 4), ohh = woff(L, 5);
    const _Float16* Wxz = p.wts + oxz;
    const _Float16* Whz = p.wts + ohz;
    const _Float16* Wxr = p.wts + oxr;
    const _Float16* Whr = p.wts + ohr;
    const _Float16* Wxh = p.wts + oxh;
    const _Float16* Whh = p.wts + ohh;
    const float* bxz = p.bias[L * 6 + 0];
    const float* bhz = p.bias[L * 6 + 1];
    const float* bxr = p.bias[L * 6 + 2];
    const float* bhr = p.bias[L * 6 + 3];
    const float* bxh = p.bias[L * 6 + 4];
    const float* bhh = p.bias[L * 6 + 5];

    const int m = lane & 15, hf = lane >> 4;

    // ---- pass 1: R = sigmoid(x Wxr + H Whr + b), stage R*H_{t-1} in LDS ----
#pragma unroll 1
    for (int nt = 0; nt < NT; ++nt) {
        int n = nt * 16 + m;
        float br = (n < dout) ? (bxr[n] + bhr[n]) : 0.0f;
        f8 acc;
#pragma unroll
        for (int j = 0; j < 8; ++j) acc[j] = br;
#pragma unroll
        for (int kt = 0; kt < KXT; ++kt)
            acc = wmma_f16(load_x_frag<L>(xg, Hw, nodeBase, p.N, lane, kt),
                           load_b(Wxr, nt, kt, KXT, lane), acc);
#pragma unroll
        for (int kt = 0; kt < KHT; ++kt)
            acc = wmma_f16(load_a_lds(Hw, HSTRIDE, hoff, kt, NP, lane),
                           load_b(Whr, nt, kt, KHT, lane), acc);
#pragma unroll
        for (int j = 0; j < 8; ++j) {
            int mm = j + 8 * hf;                      // C/D layout: row = vgpr + 8*half
            float r  = sigmoidf_(acc[j]);
            float hv = (float)Hw[mm * HSTRIDE + hoff + n];
            RHw[mm * 256 + n] = (_Float16)(r * hv);
        }
    }
    if constexpr (NP < KHp) {                          // last layer: zero K-pad of R*H
        for (int idx = lane; idx < 16 * (KHp - NP); idx += 32) {
            int node = idx / (KHp - NP);
            int f = NP + idx % (KHp - NP);
            RHw[node * 256 + f] = (_Float16)0.0f;
        }
    }

    // ---- pass 2: Z and H~, combine, stage H_new ----------------------------
#pragma unroll 1
    for (int nt = 0; nt < NT; ++nt) {
        int n = nt * 16 + m;
        float bz = (n < dout) ? (bxz[n] + bhz[n]) : 0.0f;
        float bh = (n < dout) ? (bxh[n] + bhh[n]) : 0.0f;
        f8 az, ah;
#pragma unroll
        for (int j = 0; j < 8; ++j) { az[j] = bz; ah[j] = bh; }
#pragma unroll
        for (int kt = 0; kt < KXT; ++kt) {
            h16 xa = load_x_frag<L>(xg, Hw, nodeBase, p.N, lane, kt);
            az = wmma_f16(xa, load_b(Wxz, nt, kt, KXT, lane), az);
            ah = wmma_f16(xa, load_b(Wxh, nt, kt, KXT, lane), ah);
        }
#pragma unroll
        for (int kt = 0; kt < KHT; ++kt) {
            az = wmma_f16(load_a_lds(Hw, HSTRIDE, hoff, kt, NP, lane),
                          load_b(Whz, nt, kt, KHT, lane), az);
            ah = wmma_f16(load_a_lds(RHw, 256, 0, kt, KHp, lane),
                          load_b(Whh, nt, kt, KHT, lane), ah);
        }
#pragma unroll
        for (int j = 0; j < 8; ++j) {
            int mm = j + 8 * hf;
            float z  = sigmoidf_(az[j]);
            float ht = tanhf(ah[j]);
            float hv = (float)Hw[mm * HSTRIDE + hoff + n];
            float hn = z * hv + (1.0f - z) * ht;
            if constexpr (L > 0) hn = fmaxf(hn, 0.0f); // ReLU stored into hidden state
            HNw[mm * 256 + n] = (_Float16)hn;
        }
    }
    // ---- commit H_new -> H (H_{t-1} fully consumed above) ------------------
    for (int idx = lane; idx < 16 * NP; idx += 32) {
        int node = idx / NP;                           // NP is a power of two
        int f = idx - node * NP;
        Hw[node * HSTRIDE + hoff + f] = HNw[node * 256 + f];
    }
}

__global__ __launch_bounds__(NWAVES * 32) void rgcn_kernel(Params p) {
    __shared__ __align__(16) _Float16 Hs[NWAVES * 16 * HSTRIDE]; // hidden, all 5 layers
    __shared__ __align__(16) _Float16 RHs[NWAVES * 16 * 256];    // R*H staging
    __shared__ __align__(16) _Float16 HNs[NWAVES * 16 * 256];    // H_new staging

    const int lane = threadIdx.x & 31;
    const int wave = threadIdx.x >> 5;
    _Float16* Hw  = Hs  + wave * 16 * HSTRIDE;
    _Float16* RHw = RHs + wave * 16 * 256;
    _Float16* HNw = HNs + wave * 16 * 256;
    const int nodeBase = (blockIdx.x * NWAVES + wave) * 16;

    for (int i = lane; i < 16 * HSTRIDE; i += 32) Hw[i] = (_Float16)0.0f;
    // waves are fully independent: no __syncthreads needed anywhere.

    for (int t = 0; t < T_STEPS; ++t) {
        const float* xg = p.graphs + (size_t)t * p.N * 64;
        layer_step<0>(p, xg, Hw, RHw, HNw, nodeBase, lane);
        layer_step<1>(p, xg, Hw, RHw, HNw, nodeBase, lane);
        layer_step<2>(p, xg, Hw, RHw, HNw, nodeBase, lane);
        layer_step<3>(p, xg, Hw, RHw, HNw, nodeBase, lane);
        layer_step<4>(p, xg, Hw, RHw, HNw, nodeBase, lane);
        if (lane < 16) {
            int node = nodeBase + lane;
            if (node < p.N) {
                float* o = p.out + ((size_t)t * p.N + node) * 2;
                o[0] = (float)Hw[lane * HSTRIDE + HOFF_[4] + 0];
                o[1] = (float)Hw[lane * HSTRIDE + HOFF_[4] + 1];
            }
        }
    }
}

// ---------------------------------------------------------------------------
// d_in layout (JAX pytree: dict keys sorted): [0]=graphs, [1]=edge_index,
// then per layer L (base = 2 + 12L):
//   +0 Whh +1 Whr +2 Whz +3 Wxh +4 Wxr +5 Wxz +6 bhh +7 bhr +8 bhz +9 bxh +10 bxr +11 bxz
// ---------------------------------------------------------------------------
extern "C" void kernel_launch(void* const* d_in, const int* in_sizes, int n_in,
                              void* d_out, int out_size, void* d_ws, size_t ws_size,
                              hipStream_t stream) {
    (void)n_in; (void)out_size; (void)ws_size;
    const float* graphs = (const float*)d_in[0];
    _Float16* ws16 = (_Float16*)d_ws;
    int N = in_sizes[0] / (T_STEPS * 64);

    static const int widx[6] = {5, 2, 4, 1, 3, 0};  // xz,hz,xr,hr,xh,hh -> sorted-key slot
    for (int L = 0; L < 5; ++L) {
        int base = 2 + 12 * L;
        int din = DIN_[L], dout = DOUT_[L], np = NPAD_[L], khp = KHP_[L];
        for (int mi = 0; mi < 6; ++mi) {
            bool isX = (mi == 0 || mi == 2 || mi == 4);
            int K    = isX ? din : dout;
            int Kpad = isX ? din : khp;              // all din are multiples of 32
            const float* src = (const float*)d_in[base + widx[mi]];
            _Float16* dst = ws16 + woff(L, mi);
            int total = Kpad * np;
            repack_w<<<(total + 255) / 256, 256, 0, stream>>>(src, dst, K, dout, Kpad, np);
        }
    }

    Params P;
    P.graphs = graphs;
    P.out    = (float*)d_out;
    P.wts    = ws16;
    P.N      = N;
    static const int bidx[6] = {11, 8, 10, 7, 9, 6}; // bxz,bhz,bxr,bhr,bxh,bhh
    for (int L = 0; L < 5; ++L) {
        int base = 2 + 12 * L;
        for (int mi = 0; mi < 6; ++mi)
            P.bias[L * 6 + mi] = (const float*)d_in[base + bidx[mi]];
    }

    int nodesPerBlock = NWAVES * 16;
    int grid = (N + nodesPerBlock - 1) / nodesPerBlock;
    rgcn_kernel<<<grid, NWAVES * 32, 0, stream>>>(P);
}
